// VQAE_42039139893262
// MI455X (gfx1250) — matrix operand order repaired and checked by reading, not compiled, not run on hardware
//
#include <hip/hip_runtime.h>
#include <hip/hip_bf16.h>
#include <math.h>

typedef _Float16 v16h __attribute__((ext_vector_type(16)));
typedef _Float16 h8   __attribute__((ext_vector_type(8)));
typedef float    v8f  __attribute__((ext_vector_type(8)));

#define N_  131072
#define D_  512
#define H_  128
#define E_  32
#define K_  512
#define EPS 1e-5f

// ---------------------------------------------------------------------------
// zero a small region of workspace
__global__ void k_zero(float* __restrict__ p, int n) {
    int i = blockIdx.x * 256 + threadIdx.x;
    if (i < n) p[i] = 0.f;
}

// ---------------------------------------------------------------------------
// Encoder layer 1: H1 = X @ W1 + b1  (f16 WMMA, f32 accum), plus per-column
// sum / sumsq for batch-norm stats.
// Block = 8 waves, tile 128 rows x 128 cols (full H) so X is streamed from
// HBM exactly once. Each wave owns a 32Mx64N sub-tile: 2 A-frags x 4 B-frags
// -> 8 independent v_wmma per 32-K chunk.
__global__ __launch_bounds__(256) void k_enc1(
    const float* __restrict__ X, const float* __restrict__ W1,
    const float* __restrict__ b1, float* __restrict__ H1,
    float* __restrict__ colsum, float* __restrict__ colsq)
{
    __shared__ _Float16 As[128][40];   // padded: 80B row stride, conflict-free
    __shared__ _Float16 Bs[128][40];   // W1 chunk, transposed [col][k]
    __shared__ float csum[128], csq[128];
    const int tid  = threadIdx.x;
    const int lane = tid & 31;
    const int wave = tid >> 5;
    const int r0 = blockIdx.x * 128;
    const int m    = lane & 15;
    const int hsel = (lane < 16) ? 0 : 8;
    const int am = (wave >> 1) * 32;   // wave's 32-row strip
    const int bn = (wave & 1) * 64;    // wave's 64-col strip
    if (tid < 128) { csum[tid] = 0.f; csq[tid] = 0.f; }

    v8f acc[2][4] = {};
    for (int kb = 0; kb < D_; kb += 32) {
        __syncthreads();
        // stage A: 128 rows x 32 cols of X, f32 -> f16 (coalesced float4)
#pragma unroll
        for (int i = 0; i < 4; ++i) {
            int idx = tid + i * 256;
            int row = idx >> 3, c4 = (idx & 7) << 2;
            float4 v = *(const float4*)(X + (size_t)(r0 + row) * D_ + kb + c4);
            As[row][c4 + 0] = (_Float16)v.x;
            As[row][c4 + 1] = (_Float16)v.y;
            As[row][c4 + 2] = (_Float16)v.z;
            As[row][c4 + 3] = (_Float16)v.w;
        }
        // stage B transposed: W1[kb..kb+31][0..127] -> Bs[col][k]
#pragma unroll
        for (int i = 0; i < 4; ++i) {
            int idx = tid + i * 256;
            int kr = idx >> 5, c4 = (idx & 31) << 2;
            float4 v = *(const float4*)(W1 + (size_t)(kb + kr) * H_ + c4);
            Bs[c4 + 0][kr] = (_Float16)v.x;
            Bs[c4 + 1][kr] = (_Float16)v.y;
            Bs[c4 + 2][kr] = (_Float16)v.z;
            Bs[c4 + 3][kr] = (_Float16)v.w;
        }
        __syncthreads();
        // fragments (A 16x32 / B 32x16; lane half h holds k = 16h..16h+15 contiguously in both)
        const int kB = (lane < 16) ? 0 : 16;
        v16h afr[2], bfr[4];
#pragma unroll
        for (int mt = 0; mt < 2; ++mt) {
            h8 lo = *(const h8*)&As[am + mt * 16 + m][kB];
            h8 hi = *(const h8*)&As[am + mt * 16 + m][kB + 8];
#pragma unroll
            for (int i = 0; i < 8; ++i) { afr[mt][i] = lo[i]; afr[mt][i + 8] = hi[i]; }
        }
#pragma unroll
        for (int nt = 0; nt < 4; ++nt) {
            h8 lo = *(const h8*)&Bs[bn + nt * 16 + m][kB];
            h8 hi = *(const h8*)&Bs[bn + nt * 16 + m][kB + 8];
#pragma unroll
            for (int i = 0; i < 8; ++i) { bfr[nt][i] = lo[i]; bfr[nt][i + 8] = hi[i]; }
        }
#pragma unroll
        for (int mt = 0; mt < 2; ++mt)
#pragma unroll
            for (int nt = 0; nt < 4; ++nt)
                acc[mt][nt] = __builtin_amdgcn_wmma_f32_16x16x32_f16(
                    false, afr[mt], false, bfr[nt], (short)0, acc[mt][nt], false, false);
    }
    // epilogue: bias, store H1, per-column sum/sumsq reduction
#pragma unroll
    for (int nt = 0; nt < 4; ++nt) {
        const int c = bn + nt * 16 + m;
        const float bias = b1[c];
        float s = 0.f, q = 0.f;
#pragma unroll
        for (int mt = 0; mt < 2; ++mt) {
#pragma unroll
            for (int v = 0; v < 8; ++v) {
                float h = acc[mt][nt][v] + bias;
                H1[(size_t)(r0 + am + mt * 16 + v + hsel) * H_ + c] = h;
                s += h; q += h * h;
            }
        }
        s += __shfl_xor(s, 16, 32);
        q += __shfl_xor(q, 16, 32);
        if (lane < 16) { atomicAdd(&csum[c], s); atomicAdd(&csq[c], q); }
    }
    __syncthreads();
    if (tid < 128) { atomicAdd(&colsum[tid], csum[tid]);
                     atomicAdd(&colsq [tid], csq [tid]); }
}

// ---------------------------------------------------------------------------
// finalize encoder BN affine:  scale = g*rsqrt(var+eps), shift = be - mu*scale
__global__ void k_encbn(const float* __restrict__ colsum, const float* __restrict__ colsq,
                        const float* __restrict__ g, const float* __restrict__ be,
                        float* __restrict__ scale, float* __restrict__ shift)
{
    int j = threadIdx.x;  // 128
    float mu  = colsum[j] / (float)N_;
    float var = colsq[j] / (float)N_ - mu * mu;
    float sc  = g[j] * rsqrtf(var + EPS);
    scale[j] = sc;
    shift[j] = be[j] - mu * sc;
}

// codebook row norms ||c_k||^2
__global__ void k_cn(const float* __restrict__ C, float* __restrict__ cn) {
    int k = blockIdx.x * 256 + threadIdx.x;
    if (k < K_) {
        float s = 0.f;
        const float* r = C + (size_t)k * E_;
#pragma unroll
        for (int e = 0; e < E_; ++e) s += r[e] * r[e];
        cn[k] = s;
    }
}

// ---------------------------------------------------------------------------
// Encoder layer 2: Z = relu(BN(H1)) @ W2 + b2. Tile 64x32, 8 waves (4M x 2N).
__global__ __launch_bounds__(256) void k_enc2(
    const float* __restrict__ H1, const float* __restrict__ scale,
    const float* __restrict__ shift, const float* __restrict__ W2,
    const float* __restrict__ b2, float* __restrict__ Z)
{
    __shared__ _Float16 As[64][40];
    __shared__ _Float16 Bs[32][40];
    __shared__ float ssc[128], ssh[128];
    const int tid = threadIdx.x, lane = tid & 31, wave = tid >> 5;
    const int r0 = blockIdx.x * 64;
    const int mr = (wave >> 1) * 16, nc = (wave & 1) * 16;
    const int m = lane & 15, hsel = (lane < 16) ? 0 : 8;
    if (tid < 128) { ssc[tid] = scale[tid]; ssh[tid] = shift[tid]; }

    v8f acc = {};
    for (int kb = 0; kb < H_; kb += 32) {
        __syncthreads();
#pragma unroll
        for (int i = 0; i < 2; ++i) {   // A: 64x32, fused BN+ReLU on load
            int idx = tid + i * 256;
            int row = idx >> 3, c4 = (idx & 7) << 2;
            float4 v = *(const float4*)(H1 + (size_t)(r0 + row) * H_ + kb + c4);
            int k = kb + c4;
            As[row][c4 + 0] = (_Float16)fmaxf(0.f, v.x * ssc[k+0] + ssh[k+0]);
            As[row][c4 + 1] = (_Float16)fmaxf(0.f, v.y * ssc[k+1] + ssh[k+1]);
            As[row][c4 + 2] = (_Float16)fmaxf(0.f, v.z * ssc[k+2] + ssh[k+2]);
            As[row][c4 + 3] = (_Float16)fmaxf(0.f, v.w * ssc[k+3] + ssh[k+3]);
        }
#pragma unroll
        for (int i = 0; i < 2; ++i) {   // B: W2[kb..kb+31][0..31] transposed
            int idx = tid + i * 256;
            int kr = idx >> 4, c2 = (idx & 15) << 1;
            float2 v = *(const float2*)(W2 + (size_t)(kb + kr) * E_ + c2);
            Bs[c2 + 0][kr] = (_Float16)v.x;
            Bs[c2 + 1][kr] = (_Float16)v.y;
        }
        __syncthreads();
        h8 alo = *(const h8*)&As[mr + m][(lane < 16) ? 0 : 16];
        h8 ahi = *(const h8*)&As[mr + m][((lane < 16) ? 0 : 16) + 8];
        const int kB = (lane < 16) ? 0 : 16;
        h8 blo = *(const h8*)&Bs[nc + m][kB];
        h8 bhi = *(const h8*)&Bs[nc + m][kB + 8];
        v16h a, b;
#pragma unroll
        for (int i = 0; i < 8; ++i) { a[i] = alo[i]; a[i+8] = ahi[i];
                                      b[i] = blo[i]; b[i+8] = bhi[i]; }
        acc = __builtin_amdgcn_wmma_f32_16x16x32_f16(false, a, false, b,
                                                     (short)0, acc, false, false);
    }
    float bias = b2[nc + m];
#pragma unroll
    for (int v = 0; v < 8; ++v)
        Z[(size_t)(r0 + mr + v + hsel) * E_ + nc + m] = acc[v] + bias;
}

// ---------------------------------------------------------------------------
// Distances + argmin + histogram + min-dist sum.  G = Z @ C^T via one WMMA per
// 16x16 tile (K-dim = E = 32). d = zn + cn - 2G; argmin over k independent of
// zn, so track min of (cn - 2G) and add zn at the end.
__global__ __launch_bounds__(256) void k_assign(
    const float* __restrict__ Z, const float* __restrict__ C,
    const float* __restrict__ cn, int* __restrict__ topics,
    unsigned int* __restrict__ counts, float* __restrict__ loss_acc)
{
    const int tid = threadIdx.x, lane = tid & 31, wave = tid >> 5;
    const int mr = (blockIdx.x * 8 + wave) * 16;
    const int m = lane & 15, hsel = (lane < 16) ? 0 : 8;
    const float* zr = Z + (size_t)(mr + m) * E_;

    float zn = 0.f;
#pragma unroll
    for (int e = 0; e < E_; e += 4) {
        float4 v = *(const float4*)(zr + e);
        zn += v.x*v.x + v.y*v.y + v.z*v.z + v.w*v.w;
    }
    const int kA = (lane < 16) ? 0 : 16;
    v16h a;
    {
        float4 g0 = *(const float4*)(zr + kA);
        float4 g1 = *(const float4*)(zr + kA + 4);
        float4 g2 = *(const float4*)(zr + kA + 8);
        float4 g3 = *(const float4*)(zr + kA + 12);
        a[0]=(_Float16)g0.x; a[1]=(_Float16)g0.y; a[2]=(_Float16)g0.z; a[3]=(_Float16)g0.w;
        a[4]=(_Float16)g1.x; a[5]=(_Float16)g1.y; a[6]=(_Float16)g1.z; a[7]=(_Float16)g1.w;
        a[8]=(_Float16)g2.x; a[9]=(_Float16)g2.y; a[10]=(_Float16)g2.z; a[11]=(_Float16)g2.w;
        a[12]=(_Float16)g3.x; a[13]=(_Float16)g3.y; a[14]=(_Float16)g3.z; a[15]=(_Float16)g3.w;
    }
    float minv[8]; int mini[8];
#pragma unroll
    for (int v = 0; v < 8; ++v) { minv[v] = 3.4e38f; mini[v] = 0; }

    for (int kt = 0; kt < K_; kt += 16) {
        // B = C^T fragment: col n = kt+m is codebook row (contiguous 32 f32)
        const float* cb = C + (size_t)(kt + m) * E_ + ((lane < 16) ? 0 : 16);
        float4 c0v = *(const float4*)(cb);
        float4 c1v = *(const float4*)(cb + 4);
        float4 c2v = *(const float4*)(cb + 8);
        float4 c3v = *(const float4*)(cb + 12);
        v16h b;
        b[0]=(_Float16)c0v.x; b[1]=(_Float16)c0v.y; b[2]=(_Float16)c0v.z; b[3]=(_Float16)c0v.w;
        b[4]=(_Float16)c1v.x; b[5]=(_Float16)c1v.y; b[6]=(_Float16)c1v.z; b[7]=(_Float16)c1v.w;
        b[8]=(_Float16)c2v.x; b[9]=(_Float16)c2v.y; b[10]=(_Float16)c2v.z; b[11]=(_Float16)c2v.w;
        b[12]=(_Float16)c3v.x; b[13]=(_Float16)c3v.y; b[14]=(_Float16)c3v.z; b[15]=(_Float16)c3v.w;
        v8f acc = {};
        acc = __builtin_amdgcn_wmma_f32_16x16x32_f16(false, a, false, b,
                                                     (short)0, acc, false, false);
        float cnv = cn[kt + m];
#pragma unroll
        for (int v = 0; v < 8; ++v) {
            float d = cnv - 2.f * acc[v];
            if (d < minv[v]) { minv[v] = d; mini[v] = kt + m; }
        }
    }
    // reduce argmin across the 16 lanes of each half-wave (same rows)
#pragma unroll
    for (int off = 1; off < 16; off <<= 1) {
#pragma unroll
        for (int v = 0; v < 8; ++v) {
            float ov = __shfl_xor(minv[v], off, 32);
            int   oi = __shfl_xor(mini[v], off, 32);
            if (ov < minv[v] || (ov == minv[v] && oi < mini[v])) {
                minv[v] = ov; mini[v] = oi;
            }
        }
    }
    float znr[8];
#pragma unroll
    for (int v = 0; v < 8; ++v) znr[v] = __shfl(zn, v + hsel, 32);  // uniform
    if (m == 0) {   // lanes 0 (rows 0-7) and 16 (rows 8-15) write
        float s = 0.f;
#pragma unroll
        for (int v = 0; v < 8; ++v) {
            int row = mr + v + hsel;
            topics[row] = mini[v];
            atomicAdd(&counts[mini[v]], 1u);
            s += znr[v] + minv[v];
        }
        atomicAdd(&loss_acc[0], s);
    }
}

// ---------------------------------------------------------------------------
// Tiny decoder precomputes (512 distinct rows total).
__global__ void k_dec1(const float* __restrict__ C, const float* __restrict__ W,
                       const float* __restrict__ b, float* __restrict__ Hc)
{
    int idx = blockIdx.x * 256 + threadIdx.x;  // 512*128
    int r = idx >> 7, c = idx & 127;
    float s = b[c];
#pragma unroll
    for (int e = 0; e < E_; ++e) s += C[(size_t)r * E_ + e] * W[(size_t)e * H_ + c];
    Hc[idx] = s;
}

// count-weighted BN stats over the (virtual) N-row decoder hidden activations
__global__ void k_decbn(const float* __restrict__ Hc, const unsigned int* __restrict__ counts,
                        const float* __restrict__ g, const float* __restrict__ be,
                        float* __restrict__ dscale, float* __restrict__ dshift)
{
    int j = threadIdx.x;  // 128
    float s = 0.f, q = 0.f;
    for (int k = 0; k < K_; ++k) {
        float w = (float)counts[k];
        float h = Hc[(size_t)k * H_ + j];
        s += w * h; q += w * h * h;
    }
    float mu  = s / (float)N_;
    float var = q / (float)N_ - mu * mu;
    float sc  = g[j] * rsqrtf(var + EPS);
    dscale[j] = sc;
    dshift[j] = be[j] - mu * sc;
}

__global__ void k_dec2(const float* __restrict__ Hc, const float* __restrict__ dscale,
                       const float* __restrict__ dshift, const float* __restrict__ W2,
                       const float* __restrict__ b2, float* __restrict__ Xc)
{
    int idx = blockIdx.x * 256 + threadIdx.x;  // 512*512
    int r = idx >> 9, c = idx & 511;
    float s = b2[c];
    for (int j = 0; j < H_; ++j) {
        float h = fmaxf(0.f, Hc[(size_t)r * H_ + j] * dscale[j] + dshift[j]);
        s += h * W2[(size_t)j * D_ + c];
    }
    Xc[idx] = s;
}

// ---------------------------------------------------------------------------
// Residual: sum over n of ||Xc[topics[n]] - X[n]||^2 (X: 268 MB streamed once,
// Xc 1 MB stays in L2).
__global__ __launch_bounds__(256) void k_resid(
    const float* __restrict__ X, const float* __restrict__ Xc,
    const int* __restrict__ topics, float* __restrict__ loss_acc)
{
    __shared__ float red[256];
    const int tid = threadIdx.x;
    float s = 0.f;
    const size_t total = (size_t)N_ * (D_ / 4);   // float4 units
    for (size_t i = (size_t)blockIdx.x * 256 + tid; i < total;
         i += (size_t)gridDim.x * 256) {
        int n  = (int)(i >> 7);
        int c4 = (int)(i & 127) << 2;
        int t  = topics[n];
        float4 xv = *(const float4*)(X + i * 4);
        float4 cv = *(const float4*)(Xc + (size_t)t * D_ + c4);
        float d0 = cv.x - xv.x, d1 = cv.y - xv.y;
        float d2 = cv.z - xv.z, d3 = cv.w - xv.w;
        s += d0 * d0 + d1 * d1 + d2 * d2 + d3 * d3;
    }
    red[tid] = s; __syncthreads();
    for (int o = 128; o > 0; o >>= 1) {
        if (tid < o) red[tid] += red[tid + o];
        __syncthreads();
    }
    if (tid == 0) atomicAdd(&loss_acc[1], red[0]);
}

// loss = Z_loss + C_loss + sqrt(res) = 2*minsum + sqrt(res)
__global__ void k_final(const float* __restrict__ loss_acc, float* __restrict__ out) {
    if (threadIdx.x == 0 && blockIdx.x == 0)
        out[0] = 2.f * loss_acc[0] + sqrtf(loss_acc[1]);
}

// ---------------------------------------------------------------------------
extern "C" void kernel_launch(void* const* d_in, const int* in_sizes, int n_in,
                              void* d_out, int out_size, void* d_ws, size_t ws_size,
                              hipStream_t stream) {
    (void)in_sizes; (void)n_in; (void)out_size; (void)ws_size;
    const float* X    = (const float*)d_in[0];
    const float* eW1  = (const float*)d_in[1];
    const float* eb1  = (const float*)d_in[2];
    const float* eg1  = (const float*)d_in[3];
    const float* ebe1 = (const float*)d_in[4];
    const float* eW2  = (const float*)d_in[5];
    const float* eb2  = (const float*)d_in[6];
    const float* Cb   = (const float*)d_in[7];
    const float* dW1  = (const float*)d_in[8];
    const float* db1  = (const float*)d_in[9];
    const float* dg1  = (const float*)d_in[10];
    const float* dbe1 = (const float*)d_in[11];
    const float* dW2  = (const float*)d_in[12];
    const float* db2  = (const float*)d_in[13];

    float* ws = (float*)d_ws;
    size_t off = 0;
    float* H1 = ws + off;              off += (size_t)N_ * H_;   // 64 MB
    float* Z  = ws + off;              off += (size_t)N_ * E_;   // 16 MB
    int*   topics = (int*)(ws + off);  off += N_;
    float* Hc = ws + off;              off += (size_t)K_ * H_;
    float* Xc = ws + off;              off += (size_t)K_ * D_;
    float* cn = ws + off;              off += K_;
    float* esc = ws + off;             off += H_;
    float* esh = ws + off;             off += H_;
    float* dsc = ws + off;             off += H_;
    float* dsh = ws + off;             off += H_;
    float* zreg = ws + off;            // zero-init region: 770 floats
    float* colsum = zreg;
    float* colsq  = zreg + 128;
    unsigned int* counts = (unsigned int*)(zreg + 256);
    float* lacc = zreg + 768;          // [0]=min-dist sum, [1]=residual sq

    k_zero  <<<4, 256, 0, stream>>>(zreg, 770);
    k_enc1  <<<N_ / 128, 256, 0, stream>>>(X, eW1, eb1, H1, colsum, colsq);
    k_encbn <<<1, 128, 0, stream>>>(colsum, colsq, eg1, ebe1, esc, esh);
    k_cn    <<<2, 256, 0, stream>>>(Cb, cn);
    k_enc2  <<<N_ / 64, 256, 0, stream>>>(H1, esc, esh, eW2, eb2, Z);
    k_assign<<<N_ / 128, 256, 0, stream>>>(Z, Cb, cn, topics, counts, lacc);
    k_dec1  <<<(K_ * H_) / 256, 256, 0, stream>>>(Cb, dW1, db1, Hc);
    k_decbn <<<1, 128, 0, stream>>>(Hc, counts, dg1, dbe1, dsc, dsh);
    k_dec2  <<<(K_ * D_) / 256, 256, 0, stream>>>(Hc, dsc, dsh, dW2, db2, Xc);
    k_resid <<<4096, 256, 0, stream>>>(X, Xc, topics, lacc);
    k_final <<<1, 1, 0, stream>>>(lacc, (float*)d_out);
}
